// MPCN_42236708389405
// MI455X (gfx1250) — matrix operand order repaired and checked by reading, not compiled, not run on hardware
//
#include <hip/hip_runtime.h>
#include <hip/hip_bf16.h>
#include <math.h>

#define Bc 512
#define Rc 20
#define Lc 50
#define Dc 64
#define Pc 2
#define Kc 10
#define BR (Bc * Rc)
#define IN_DIMc (2 * Pc * Dc)   // 256

#if defined(__has_builtin)
#if __has_builtin(__builtin_amdgcn_global_load_async_to_lds_b32) && \
    __has_builtin(__builtin_amdgcn_s_wait_asynccnt)
#define HAVE_ASYNC_LDS 1
#endif
#endif

typedef __attribute__((ext_vector_type(16))) _Float16 v16h;
typedef __attribute__((ext_vector_type(8)))  float    v8f;
typedef __attribute__((ext_vector_type(8)))  _Float16 h8;

#define WMMA_F16(a, b, c) \
  __builtin_amdgcn_wmma_f32_16x16x32_f16(false, (a), false, (b), (short)0, (c), false, false)

// ---- fast device math (native v_exp_f32 / v_rcp_f32 paths) -----------------
__device__ __forceinline__ float fast_sigmoid(float x) {
  return __builtin_amdgcn_rcpf(1.f + __expf(-x));
}
__device__ __forceinline__ float fast_tanh(float x) {
  return 1.f - 2.f * __builtin_amdgcn_rcpf(__expf(2.f * x) + 1.f);
}

// ---- WMMA operand loaders (LDS f16, row-major, stride in halves) ----------
// A operand: 16x32 tile, rows row0..row0+15, K columns kb..kb+31.
// Lane l<16 holds row (row0+l), K = kb+{0..7, 16..23}; lane l>=16 holds
// row (row0+l-16), K = kb+{8..15, 24..31}.  (ISA 7.12.2, 16-bit A 16x32)
__device__ __forceinline__ v16h ld_matA(const _Float16* m, int stride, int row0,
                                        int kb, int lane) {
  int r = row0 + (lane & 15);
  int o = (lane & 16) ? 8 : 0;
  const _Float16* p = m + r * stride + kb + o;
  h8 lo = *(const h8*)p;
  h8 hi = *(const h8*)(p + 16);
  v16h a;
#pragma unroll
  for (int i = 0; i < 8; ++i) { a[i] = lo[i]; a[i + 8] = hi[i]; }
  return a;
}

// B operand from "Bt" stored row-major as Bt[n][k] (i.e. B(k,n) = Bt[n][k]).
// Lane l<16 holds col (col0+l), K = kb+{0..7, 16..23}; lane l>=16 holds col
// (col0+l-16), K = kb+{8..15, 24..31}.
__device__ __forceinline__ v16h ld_matB(const _Float16* m, int stride, int col0,
                                        int kb, int lane) {
  int c = col0 + (lane & 15);
  int o = (lane & 16) ? 8 : 0;
  const _Float16* p = m + c * stride + kb + o;
  h8 lo = *(const h8*)p;
  h8 hi = *(const h8*)(p + 16);
  v16h b;
#pragma unroll
  for (int i = 0; i < 8; ++i) { b[i] = lo[i]; b[i + 8] = hi[i]; }
  return b;
}

// Deterministic stateless gumbel noise (stand-in for jax.random.gumbel).
__device__ __forceinline__ float gumbel_hash(unsigned a, unsigned b, unsigned c) {
  unsigned x = a * 0x9E3779B1u + b * 0x85EBCA77u + c * 0xC2B2AE3Du + 0x27D4EB2Fu;
  x ^= x >> 16; x *= 0x7FEB352Du; x ^= x >> 15; x *= 0x846CA68Bu; x ^= x >> 16;
  float u = ((float)(x >> 8) + 0.5f) * (1.0f / 16777216.0f);
  return -__logf(-__logf(u));
}

// ---- Stage 1: gather + sum over L tokens -> rev[2*BR][64] ------------------
// Uses gfx1250 async global->LDS DMA (ASYNCcnt-tracked): all 50 indexed rows
// are issued as per-lane async copies, then one s_wait_asynccnt 0 per wave.
__global__ __launch_bounds__(64) void gather_sum_kernel(
    float* __restrict__ rev, const int* __restrict__ urev,
    const int* __restrict__ irev, const float* __restrict__ emb) {
  int n = blockIdx.x;  // 0 .. 2*BR-1
  const int* src = (n < BR) ? urev : irev;
  int br = (n < BR) ? n : n - BR;
  __shared__ int idx[Lc];
  int tid = threadIdx.x;  // 64 threads = 2 waves
  if (tid < Lc) idx[tid] = src[br * Lc + tid];
  __syncthreads();
#if HAVE_ASYNC_LDS
  __shared__ __align__(16) float buf[Lc][Dc];
#pragma unroll 2
  for (int l = 0; l < Lc; ++l) {
    const float* g = emb + (size_t)idx[l] * Dc + tid;
    __builtin_amdgcn_global_load_async_to_lds_b32((int*)g, (int*)&buf[l][tid], 0, 0);
  }
  __builtin_amdgcn_s_wait_asynccnt(0);  // each lane consumes only its own slots
  asm volatile("" ::: "memory");
  float acc = 0.f;
#pragma unroll 5
  for (int l = 0; l < Lc; ++l) acc += buf[l][tid];
#else
  float acc = 0.f;
#pragma unroll 2
  for (int l = 0; l < Lc; ++l) {
    if (l + 1 < Lc) __builtin_prefetch(emb + (size_t)idx[l + 1] * Dc, 0, 1);
    acc += emb[(size_t)idx[l] * Dc + tid];
  }
#endif
  rev[(size_t)n * Dc + tid] = acc;
}

// ---- Stage 2: gating (in place), rev = sigmoid(x W1^T) + bg*tanh(x W2^T+b2)
__global__ __launch_bounds__(128) void gating_kernel(
    float* __restrict__ rev, const float* __restrict__ w1,
    const float* __restrict__ w2, const float* __restrict__ b2,
    const float* __restrict__ bg) {
  __shared__ __align__(16) _Float16 W1[64][64];
  __shared__ __align__(16) _Float16 W2[64][64];
  __shared__ __align__(16) _Float16 X[64][64];
  int tid = threadIdx.x;
  int row_base = blockIdx.x * 64;
  for (int i = tid; i < 64 * 64; i += 128) {
    W1[i >> 6][i & 63] = (_Float16)w1[i];
    W2[i >> 6][i & 63] = (_Float16)w2[i];
    X[i >> 6][i & 63] = (_Float16)rev[(size_t)(row_base + (i >> 6)) * 64 + (i & 63)];
  }
  __syncthreads();
  int wave = tid >> 5, lane = tid & 31;
  int m0 = wave * 16;
  v16h a0 = ld_matA(&X[0][0], 64, m0, 0, lane);
  v16h a1 = ld_matA(&X[0][0], 64, m0, 32, lane);
  for (int nt = 0; nt < 4; ++nt) {
    v8f acc1 = {}; v8f acc2 = {};
    v16h b0 = ld_matB(&W1[0][0], 64, nt * 16, 0, lane);
    v16h b1 = ld_matB(&W1[0][0], 64, nt * 16, 32, lane);
    acc1 = WMMA_F16(a0, b0, acc1);
    acc1 = WMMA_F16(a1, b1, acc1);
    b0 = ld_matB(&W2[0][0], 64, nt * 16, 0, lane);
    b1 = ld_matB(&W2[0][0], 64, nt * 16, 32, lane);
    acc2 = WMMA_F16(a0, b0, acc2);
    acc2 = WMMA_F16(a1, b1, acc2);
    int col = nt * 16 + (lane & 15);
    float bgv = bg[col], b2v = b2[col];
    int r0 = m0 + ((lane & 16) ? 8 : 0);
#pragma unroll
    for (int v = 0; v < 8; ++v) {
      float g = fast_sigmoid(acc1[v]) + bgv * fast_tanh(acc2[v] + b2v);
      rev[(size_t)(row_base + r0 + v) * 64 + col] = g;
    }
  }
}

// ---- Stage 3: review-level co-attention + gumbel-hard pointer selection ----
__global__ __launch_bounds__(128) void review_coatt_kernel(
    const float* __restrict__ rev, const float* __restrict__ ra_w,
    const float* __restrict__ ra_b, const float* __restrict__ ra_M,
    int* __restrict__ pr_idx, int* __restrict__ pc_idx) {
  __shared__ __align__(16) _Float16 RW[64][64];
  __shared__ __align__(16) _Float16 MT[64][64];   // MT[e][d] = M[d][e]
  __shared__ __align__(16) _Float16 U[32][64];
  __shared__ __align__(16) _Float16 I[32][64];
  __shared__ __align__(16) _Float16 UO[32][64];
  __shared__ __align__(16) _Float16 IO[32][64];
  __shared__ __align__(16) _Float16 T[32][64];
  __shared__ float S[32][32];
  __shared__ float rm[32], cm[32];

  int b = blockIdx.x;
  int tid = threadIdx.x;
  for (int i = tid; i < 64 * 64; i += 128) {
    int d = i >> 6, e = i & 63;
    RW[d][e] = (_Float16)ra_w[i];
    MT[e][d] = (_Float16)ra_M[i];
  }
  for (int i = tid; i < 32 * 64; i += 128) {
    int r = i >> 6, d = i & 63;
    float uv = (r < Rc) ? rev[(size_t)(b * Rc + r) * 64 + d] : 0.f;
    float iv = (r < Rc) ? rev[(size_t)(BR + b * Rc + r) * 64 + d] : 0.f;
    U[r][d] = (_Float16)uv;
    I[r][d] = (_Float16)iv;
  }
  __syncthreads();
  int wave = tid >> 5, lane = tid & 31;
  {  // uo / io projections (relu(x @ ra_w^T + ra_b))
    const _Float16* src = (wave < 2) ? &U[0][0] : &I[0][0];
    _Float16(*dst)[64] = (wave < 2) ? UO : IO;
    int m0 = (wave & 1) * 16;
    v16h a0 = ld_matA(src, 64, m0, 0, lane);
    v16h a1 = ld_matA(src, 64, m0, 32, lane);
    for (int nt = 0; nt < 4; ++nt) {
      v8f acc = {};
      v16h b0 = ld_matB(&RW[0][0], 64, nt * 16, 0, lane);
      v16h b1 = ld_matB(&RW[0][0], 64, nt * 16, 32, lane);
      acc = WMMA_F16(a0, b0, acc);
      acc = WMMA_F16(a1, b1, acc);
      int col = nt * 16 + (lane & 15);
      float bias = ra_b[col];
      int r0 = m0 + ((lane & 16) ? 8 : 0);
#pragma unroll
      for (int v = 0; v < 8; ++v) {
        float o = acc[v] + bias;
        dst[r0 + v][col] = (_Float16)(o > 0.f ? o : 0.f);
      }
    }
  }
  __syncthreads();
  {  // T = UO @ M  (Bt = MT)
    int m0 = (wave >> 1) * 16;
    int ntb = (wave & 1) * 2;
    v16h a0 = ld_matA(&UO[0][0], 64, m0, 0, lane);
    v16h a1 = ld_matA(&UO[0][0], 64, m0, 32, lane);
    for (int q = 0; q < 2; ++q) {
      int nt = ntb + q;
      v8f acc = {};
      v16h b0 = ld_matB(&MT[0][0], 64, nt * 16, 0, lane);
      v16h b1 = ld_matB(&MT[0][0], 64, nt * 16, 32, lane);
      acc = WMMA_F16(a0, b0, acc);
      acc = WMMA_F16(a1, b1, acc);
      int col = nt * 16 + (lane & 15);
      int r0 = m0 + ((lane & 16) ? 8 : 0);
#pragma unroll
      for (int v = 0; v < 8; ++v) T[r0 + v][col] = (_Float16)acc[v];
    }
  }
  __syncthreads();
  {  // S = T @ IO^T  (Bt = IO directly)
    int m0 = (wave >> 1) * 16;
    int nt = wave & 1;
    v16h a0 = ld_matA(&T[0][0], 64, m0, 0, lane);
    v16h a1 = ld_matA(&T[0][0], 64, m0, 32, lane);
    v16h b0 = ld_matB(&IO[0][0], 64, nt * 16, 0, lane);
    v16h b1 = ld_matB(&IO[0][0], 64, nt * 16, 32, lane);
    v8f acc = {};
    acc = WMMA_F16(a0, b0, acc);
    acc = WMMA_F16(a1, b1, acc);
    int col = nt * 16 + (lane & 15);
    int r0 = m0 + ((lane & 16) ? 8 : 0);
#pragma unroll
    for (int v = 0; v < 8; ++v) S[r0 + v][col] = acc[v];
  }
  __syncthreads();
  if (tid < Rc) {  // row max (pointer over user reviews)
    float mx = -1e30f;
    for (int c = 0; c < Rc; ++c) mx = fmaxf(mx, S[tid][c]);
    rm[tid] = mx;
  } else if (tid >= 32 && tid < 32 + Rc) {  // col max (item reviews)
    int c = tid - 32;
    float mx = -1e30f;
    for (int r = 0; r < Rc; ++r) mx = fmaxf(mx, S[r][c]);
    cm[c] = mx;
  }
  __syncthreads();
  if (tid < 2 * Pc) {  // straight-through hard one-hot == argmax(logit+gumbel)
    int p = tid >> 1, isC = tid & 1;
    const float* srcv = isC ? cm : rm;
    float best = -1e30f; int bi = 0;
    for (int r = 0; r < Rc; ++r) {
      float v = srcv[r] + gumbel_hash((unsigned)b, (unsigned)(2 * p + isC), (unsigned)r);
      if (v > best) { best = v; bi = r; }
    }
    if (isC) pc_idx[b * Pc + p] = bi; else pr_idx[b * Pc + p] = bi;
  }
}

// ---- Stage 4: word-level co-attention per (batch, pointer) -----------------
struct __align__(16) WordSmem {
  _Float16 UR[64][64];
  _Float16 IR[64][64];
  union Alias {
    struct PW { _Float16 WW[64][64]; _Float16 UO[64][64]; } pw;  // dead after T
    float SW[64][64];                                            // used after
  } a;
  _Float16 MT[64][64];
  _Float16 IO[64][64];
  _Float16 T[64][64];
  float rm[64], cm[64], wr[64], wc[64];
};

__global__ __launch_bounds__(128) void word_coatt_kernel(
    const int* __restrict__ user_reviews, const int* __restrict__ item_reviews,
    const float* __restrict__ emb, const float* __restrict__ wa_w,
    const float* __restrict__ wa_b, const float* __restrict__ wa_M,
    const int* __restrict__ pr_idx, const int* __restrict__ pc_idx,
    float* __restrict__ xbuf) {
  __shared__ WordSmem sm;
  int bp = blockIdx.x;
  int b = bp / Pc, p = bp % Pc;
  int tid = threadIdx.x;
  int ru = pr_idx[bp], ri = pc_idx[bp];
  const int* uix = user_reviews + (size_t)(b * Rc + ru) * Lc;
  const int* iix = item_reviews + (size_t)(b * Rc + ri) * Lc;
  for (int i = tid; i < 64 * 64; i += 128) {
    int d = i >> 6, e = i & 63;
    sm.a.pw.WW[d][e] = (_Float16)wa_w[i];
    sm.MT[e][d] = (_Float16)wa_M[i];
  }
  for (int i = tid; i < 64 * 64; i += 128) {
    int l = i >> 6, d = i & 63;
    float uv = 0.f, iv = 0.f;
    if (l < Lc) {
      uv = emb[(size_t)uix[l] * Dc + d];
      iv = emb[(size_t)iix[l] * Dc + d];
    }
    sm.UR[l][d] = (_Float16)uv;
    sm.IR[l][d] = (_Float16)iv;
  }
  __syncthreads();
  int wave = tid >> 5, lane = tid & 31;
  int m0 = wave * 16;
  {  // UO = relu(UR W^T + b), IO = relu(IR W^T + b)
    v16h a0 = ld_matA(&sm.UR[0][0], 64, m0, 0, lane);
    v16h a1 = ld_matA(&sm.UR[0][0], 64, m0, 32, lane);
    v16h c0 = ld_matA(&sm.IR[0][0], 64, m0, 0, lane);
    v16h c1 = ld_matA(&sm.IR[0][0], 64, m0, 32, lane);
    for (int nt = 0; nt < 4; ++nt) {
      v16h b0 = ld_matB(&sm.a.pw.WW[0][0], 64, nt * 16, 0, lane);
      v16h b1 = ld_matB(&sm.a.pw.WW[0][0], 64, nt * 16, 32, lane);
      v8f au = {}; v8f ai = {};
      au = WMMA_F16(a0, b0, au); au = WMMA_F16(a1, b1, au);
      ai = WMMA_F16(c0, b0, ai); ai = WMMA_F16(c1, b1, ai);
      int col = nt * 16 + (lane & 15);
      float bias = wa_b[col];
      int r0 = m0 + ((lane & 16) ? 8 : 0);
#pragma unroll
      for (int v = 0; v < 8; ++v) {
        float ou = au[v] + bias, oi = ai[v] + bias;
        sm.a.pw.UO[r0 + v][col] = (_Float16)(ou > 0.f ? ou : 0.f);
        sm.IO[r0 + v][col] = (_Float16)(oi > 0.f ? oi : 0.f);
      }
    }
  }
  __syncthreads();
  {  // T = UO @ M
    v16h a0 = ld_matA(&sm.a.pw.UO[0][0], 64, m0, 0, lane);
    v16h a1 = ld_matA(&sm.a.pw.UO[0][0], 64, m0, 32, lane);
    for (int nt = 0; nt < 4; ++nt) {
      v16h b0 = ld_matB(&sm.MT[0][0], 64, nt * 16, 0, lane);
      v16h b1 = ld_matB(&sm.MT[0][0], 64, nt * 16, 32, lane);
      v8f acc = {};
      acc = WMMA_F16(a0, b0, acc); acc = WMMA_F16(a1, b1, acc);
      int col = nt * 16 + (lane & 15);
      int r0 = m0 + ((lane & 16) ? 8 : 0);
#pragma unroll
      for (int v = 0; v < 8; ++v) sm.T[r0 + v][col] = (_Float16)acc[v];
    }
  }
  __syncthreads();
  {  // SW = T @ IO^T  (SW aliases WW/UO, both dead now)
    v16h a0 = ld_matA(&sm.T[0][0], 64, m0, 0, lane);
    v16h a1 = ld_matA(&sm.T[0][0], 64, m0, 32, lane);
    for (int nt = 0; nt < 4; ++nt) {
      v16h b0 = ld_matB(&sm.IO[0][0], 64, nt * 16, 0, lane);
      v16h b1 = ld_matB(&sm.IO[0][0], 64, nt * 16, 32, lane);
      v8f acc = {};
      acc = WMMA_F16(a0, b0, acc); acc = WMMA_F16(a1, b1, acc);
      int col = nt * 16 + (lane & 15);
      int r0 = m0 + ((lane & 16) ? 8 : 0);
#pragma unroll
      for (int v = 0; v < 8; ++v) sm.a.SW[r0 + v][col] = acc[v];
    }
  }
  __syncthreads();
  if (tid < Lc) {  // row means over valid 50 columns
    float s = 0.f;
    for (int j = 0; j < Lc; ++j) s += sm.a.SW[tid][j];
    sm.rm[tid] = s * (1.f / (float)Lc);
  } else if (tid >= 64 && tid < 64 + Lc) {  // col means over valid 50 rows
    int c = tid - 64;
    float s = 0.f;
    for (int l = 0; l < Lc; ++l) s += sm.a.SW[l][c];
    sm.cm[c] = s * (1.f / (float)Lc);
  }
  __syncthreads();
  if (tid < Lc) {
    float mx = -1e30f;
    for (int j = 0; j < Lc; ++j) mx = fmaxf(mx, sm.rm[j]);
    sm.wr[tid] = __expf(sm.rm[tid] - mx);
  } else if (tid >= 64 && tid < 64 + Lc) {
    int c = tid - 64;
    float mx = -1e30f;
    for (int j = 0; j < Lc; ++j) mx = fmaxf(mx, sm.cm[j]);
    sm.wc[c] = __expf(sm.cm[c] - mx);
  }
  __syncthreads();
  if (tid < 64) {  // u_repr = softmax(rowmean)^T UR
    float s = 0.f, acc = 0.f;
    for (int l = 0; l < Lc; ++l) s += sm.wr[l];
    for (int l = 0; l < Lc; ++l) acc += sm.wr[l] * (float)sm.UR[l][tid];
    xbuf[(size_t)b * IN_DIMc + p * Dc + tid] = acc * __builtin_amdgcn_rcpf(s);
  } else {         // i_repr = softmax(colmean)^T IR
    int d = tid - 64;
    float s = 0.f, acc = 0.f;
    for (int l = 0; l < Lc; ++l) s += sm.wc[l];
    for (int l = 0; l < Lc; ++l) acc += sm.wc[l] * (float)sm.IR[l][d];
    xbuf[(size_t)b * IN_DIMc + (Pc + p) * Dc + d] = acc * __builtin_amdgcn_rcpf(s);
  }
}

// ---- Stage 5: factorization machine head -----------------------------------
__global__ __launch_bounds__(256) void fm_kernel(
    const float* __restrict__ xbuf, const float* __restrict__ fm_v,
    const float* __restrict__ fm_w, const float* __restrict__ fm_b,
    float* __restrict__ out) {
  int b = blockIdx.x * blockDim.x + threadIdx.x;
  if (b >= Bc) return;
  const float* x = xbuf + (size_t)b * IN_DIMc;
  float lin = fm_b[0];
  float i1[Kc], i2[Kc];
#pragma unroll
  for (int k = 0; k < Kc; ++k) { i1[k] = 0.f; i2[k] = 0.f; }
  for (int t = 0; t < IN_DIMc; ++t) {
    float xv = x[t];
    lin += xv * fm_w[t];
    float x2 = xv * xv;
#pragma unroll
    for (int k = 0; k < Kc; ++k) {
      float v = fm_v[t * Kc + k];
      i1[k] += xv * v;
      i2[k] += x2 * v * v;
    }
  }
  float pair = 0.f;
#pragma unroll
  for (int k = 0; k < Kc; ++k) pair += i1[k] * i1[k] - i2[k];
  out[b] = lin + 0.5f * pair;
}

extern "C" void kernel_launch(void* const* d_in, const int* in_sizes, int n_in,
                              void* d_out, int out_size, void* d_ws, size_t ws_size,
                              hipStream_t stream) {
  (void)in_sizes; (void)n_in; (void)out_size; (void)ws_size;
  const int*   user_reviews = (const int*)d_in[0];
  const int*   item_reviews = (const int*)d_in[1];
  const float* emb     = (const float*)d_in[2];
  const float* gate_w1 = (const float*)d_in[3];
  const float* gate_w2 = (const float*)d_in[4];
  const float* gate_b2 = (const float*)d_in[5];
  const float* gate_bg = (const float*)d_in[6];
  const float* ra_w    = (const float*)d_in[7];
  const float* ra_b    = (const float*)d_in[8];
  const float* ra_M    = (const float*)d_in[9];
  const float* wa_w    = (const float*)d_in[10];
  const float* wa_b    = (const float*)d_in[11];
  const float* wa_M    = (const float*)d_in[12];
  const float* fm_v    = (const float*)d_in[13];
  const float* fm_w    = (const float*)d_in[14];
  const float* fm_b    = (const float*)d_in[15];
  float* out = (float*)d_out;

  // workspace layout (floats): rev[2*BR*64] | xbuf[B*256] | pr_idx | pc_idx
  float* rev  = (float*)d_ws;
  float* xbuf = rev + (size_t)2 * BR * 64;
  int*   pr_idx = (int*)(xbuf + (size_t)Bc * IN_DIMc);
  int*   pc_idx = pr_idx + Bc * Pc;

  gather_sum_kernel<<<2 * BR, 64, 0, stream>>>(rev, user_reviews, item_reviews, emb);
  gating_kernel<<<(2 * BR) / 64, 128, 0, stream>>>(rev, gate_w1, gate_w2, gate_b2, gate_bg);
  review_coatt_kernel<<<Bc, 128, 0, stream>>>(rev, ra_w, ra_b, ra_M, pr_idx, pc_idx);
  word_coatt_kernel<<<Bc * Pc, 128, 0, stream>>>(user_reviews, item_reviews, emb,
                                                 wa_w, wa_b, wa_M, pr_idx, pc_idx, xbuf);
  fm_kernel<<<(Bc + 255) / 256, 256, 0, stream>>>(xbuf, fm_v, fm_w, fm_b, out);
}
